// SAGEConvNet_12189117186688
// MI455X (gfx1250) — compile-verified, matrix operand order checked by reading, not run on hardware
//
#include <hip/hip_runtime.h>
#include <hip/hip_bf16.h>
#include <math.h>

typedef __attribute__((ext_vector_type(16))) _Float16 v16h;
typedef __attribute__((ext_vector_type(8)))  float    v8f;

#define NB_  64
#define NN_  1024
#define NE_  1048576
#define M1_  65536      /* B*N */
#define K1_  820        /* ceil(0.8*1024) */
#define M2_  52480      /* B*K1 */
#define K2_  656        /* ceil(0.8*820) */
#define M3_  41984      /* B*K2 */

// ---------------- edge scatter (segment sums) ----------------
__global__ void __launch_bounds__(256)
scatter1_kernel(const float* __restrict__ x, const int* __restrict__ src,
                const int* __restrict__ dst, float* __restrict__ ssum,
                float* __restrict__ deg, int nE) {
    int e = blockIdx.x * 256 + threadIdx.x;
    if (e >= nE) return;
    int s = src[e], d = dst[e];
    atomicAdd(&deg[d], 1.0f);
    const float* xs = x + (size_t)s * 64;
    float* ss = ssum + (size_t)d * 64;
#pragma unroll 4
    for (int f = 0; f < 64; ++f) atomicAdd(&ss[f], xs[f]);
}

__global__ void __launch_bounds__(256)
scatter2_kernel(const float* __restrict__ x, const int* __restrict__ src,
                const int* __restrict__ dst, const float* __restrict__ em,
                float* __restrict__ ssum, float* __restrict__ deg, int nE) {
    int e = blockIdx.x * 256 + threadIdx.x;
    if (e >= nE) return;
    float m = em[e];
    if (m <= 0.0f) return;
    int s = src[e], d = dst[e];
    atomicAdd(&deg[d], m);
    const float* xs = x + (size_t)s * 64;
    float* ss = ssum + (size_t)d * 64;
#pragma unroll 4
    for (int f = 0; f < 64; ++f) atomicAdd(&ss[f], xs[f] * m);
}

// ---------------- WMMA SAGEConv: relu(mean@Wl + b + x@Wr) ----------------
__device__ inline v16h frag_from_f32(const float* __restrict__ row, int kbase,
                                     int half16, float scale) {
    // 16-bit A layout: lanes 0-15 K = kbase+{0..7, 16..23}; lanes 16-31 +8
    v16h a;
    const float* p0 = row + kbase + 8 * half16;
    const float* p1 = row + kbase + 16 + 8 * half16;
#pragma unroll
    for (int t = 0; t < 8; ++t) {
        a[t]     = (_Float16)(p0[t] * scale);
        a[8 + t] = (_Float16)(p1[t] * scale);
    }
    return a;
}

__device__ inline v16h frag_from_lds(const _Float16* __restrict__ row, int kbase,
                                     int half16) {
    v16h b;
    const _Float16* p0 = row + kbase + 8 * half16;
    const _Float16* p1 = row + kbase + 16 + 8 * half16;
#pragma unroll
    for (int t = 0; t < 8; ++t) { b[t] = p0[t]; b[8 + t] = p1[t]; }
    return b;
}

__global__ void __launch_bounds__(256)
conv_wmma_kernel(const float* __restrict__ ssum, const float* __restrict__ deg,
                 const float* __restrict__ xin,
                 const float* __restrict__ Wl, const float* __restrict__ bl,
                 const float* __restrict__ Wr, float* __restrict__ hout, int M) {
    __shared__ _Float16 WlT[64 * 64];
    __shared__ _Float16 WrT[64 * 64];
    int tid = threadIdx.x;
    for (int i = tid; i < 4096; i += 256) {
        int n = i >> 6, k = i & 63;
        WlT[i] = (_Float16)Wl[k * 64 + n];   // transpose: WT[n][k]
        WrT[i] = (_Float16)Wr[k * 64 + n];
    }
    __syncthreads();

    int wave = tid >> 5, lane = tid & 31;
    int row0 = (blockIdx.x * 8 + wave) * 16;
    if (row0 >= M) return;
    int half16 = lane >> 4;
    int m = lane & 15;
    int row = row0 + m;

    float d = deg[row];
    float inv = 1.0f / (d < 1.0f ? 1.0f : d);
    const float* srow = ssum + (size_t)row * 64;
    const float* xrow = xin + (size_t)row * 64;

    v16h am0 = frag_from_f32(srow, 0, half16, inv);
    v16h am1 = frag_from_f32(srow, 32, half16, inv);
    v16h ax0 = frag_from_f32(xrow, 0, half16, 1.0f);
    v16h ax1 = frag_from_f32(xrow, 32, half16, 1.0f);

    int n = lane & 15;
#pragma unroll
    for (int nb = 0; nb < 4; ++nb) {
        const _Float16* wl_row = WlT + (nb * 16 + n) * 64;
        const _Float16* wr_row = WrT + (nb * 16 + n) * 64;
        v16h bl0 = frag_from_lds(wl_row, 0, half16);
        v16h bl1 = frag_from_lds(wl_row, 32, half16);
        v16h br0 = frag_from_lds(wr_row, 0, half16);
        v16h br1 = frag_from_lds(wr_row, 32, half16);
        v8f acc = {};
        acc = __builtin_amdgcn_wmma_f32_16x16x32_f16(false, am0, false, bl0,
                                                     (short)0, acc, false, false);
        acc = __builtin_amdgcn_wmma_f32_16x16x32_f16(false, am1, false, bl1,
                                                     (short)0, acc, false, false);
        acc = __builtin_amdgcn_wmma_f32_16x16x32_f16(false, ax0, false, br0,
                                                     (short)0, acc, false, false);
        acc = __builtin_amdgcn_wmma_f32_16x16x32_f16(false, ax1, false, br1,
                                                     (short)0, acc, false, false);
        float bias = bl[nb * 16 + n];
#pragma unroll
        for (int v = 0; v < 8; ++v) {
            float val = acc[v] + bias;
            val = val > 0.0f ? val : 0.0f;   // relu
            hout[(size_t)(row0 + v + 8 * half16) * 64 + nb * 16 + n] = val;
        }
    }
}

// ---------------- score = (h . p) / ||p|| ----------------
__global__ void __launch_bounds__(256)
score_kernel(const float* __restrict__ h, const float* __restrict__ p,
             float* __restrict__ score, int M) {
    int i = blockIdx.x * 256 + threadIdx.x;
    if (i >= M) return;
    const float* row = h + (size_t)i * 64;
    float nrm = 0.0f, s = 0.0f;
#pragma unroll 8
    for (int f = 0; f < 64; ++f) { float pv = p[f]; nrm += pv * pv; s += row[f] * pv; }
    score[i] = s / sqrtf(nrm);
}

// ---------------- per-graph top-k via bitonic sort (descending) ----------------
__global__ void __launch_bounds__(256)
topk_kernel(const float* __restrict__ score, int n_per, int n_pad, int k,
            int* __restrict__ glob, float* __restrict__ gate,
            int* __restrict__ mapping) {
    __shared__ float sval[1024];
    __shared__ int   sidx[1024];
    int g = blockIdx.x, tid = threadIdx.x;
    for (int j = tid; j < n_pad; j += 256) {
        sval[j] = (j < n_per) ? score[g * n_per + j] : -3.4e38f;
        sidx[j] = j;
    }
    __syncthreads();
    for (int sz = 2; sz <= n_pad; sz <<= 1) {
        for (int st = sz >> 1; st > 0; st >>= 1) {
            for (int j = tid; j < n_pad; j += 256) {
                int l = j ^ st;
                if (l > j) {
                    bool desc = ((j & sz) == 0);
                    float a = sval[j], b = sval[l];
                    if (desc ? (a < b) : (a > b)) {
                        int ia = sidx[j], ib = sidx[l];
                        sval[j] = b; sval[l] = a;
                        sidx[j] = ib; sidx[l] = ia;
                    }
                }
            }
            __syncthreads();
        }
    }
    for (int r = tid; r < k; r += 256) {
        int gl = g * n_per + sidx[r];
        int nj = g * k + r;
        glob[nj] = gl;
        gate[nj] = tanhf(sval[r]);
        mapping[gl] = nj;
    }
}

// ---------------- x_new = h[glob] * gate ----------------
__global__ void __launch_bounds__(256)
gather_kernel(const float* __restrict__ h, const int* __restrict__ glob,
              const float* __restrict__ gate, float* __restrict__ xnew, int Mk) {
    int t = blockIdx.x * 256 + threadIdx.x;
    int node = t >> 4, q = t & 15;
    if (node >= Mk) return;
    float ga = gate[node];
    int gl = glob[node];
    float4 v = ((const float4*)(h + (size_t)gl * 64))[q];
    v.x *= ga; v.y *= ga; v.z *= ga; v.w *= ga;
    ((float4*)(xnew + (size_t)node * 64))[q] = v;
}

// ---------------- edge remap through mapping ----------------
__global__ void __launch_bounds__(256)
remap_kernel(const int* __restrict__ ei, const int* __restrict__ mapping,
             int* __restrict__ src2, int* __restrict__ dst2,
             float* __restrict__ em2, int nE) {
    int e = blockIdx.x * 256 + threadIdx.x;
    if (e >= nE) return;
    int s = mapping[ei[e]];
    int d = mapping[ei[nE + e]];
    bool keep = (s >= 0) && (d >= 0);
    src2[e] = s >= 0 ? s : 0;
    dst2[e] = d >= 0 ? d : 0;
    em2[e] = keep ? 1.0f : 0.0f;
}

// ---------------- per-graph [max || mean] readout ----------------
__global__ void __launch_bounds__(64)
readout_kernel(const float* __restrict__ xnew, int k, float* __restrict__ xout) {
    int g = blockIdx.x, f = threadIdx.x;
    const float* base = xnew + (size_t)g * k * 64 + f;
    float mx = -3.4e38f, sm = 0.0f;
    for (int r = 0; r < k; ++r) {
        float v = base[(size_t)r * 64];
        mx = v > mx ? v : mx;
        sm += v;
    }
    xout[g * 128 + f] = mx;
    xout[g * 128 + 64 + f] = sm / (float)k;
}

// ---------------- final MLP: relu((x1+x2)@Wf1+bf1)@Wf2+bf2 ----------------
__global__ void __launch_bounds__(64)
mlp_kernel(const float* __restrict__ x1, const float* __restrict__ x2,
           const float* __restrict__ Wf1, const float* __restrict__ bf1,
           const float* __restrict__ Wf2, const float* __restrict__ bf2,
           float* __restrict__ out) {
    __shared__ float z[128];
    __shared__ float z1[64];
    int g = blockIdx.x, j = threadIdx.x;
    z[j]      = x1[g * 128 + j]      + x2[g * 128 + j];
    z[64 + j] = x1[g * 128 + 64 + j] + x2[g * 128 + 64 + j];
    __syncthreads();
    float acc = bf1[j];
#pragma unroll 8
    for (int kk = 0; kk < 128; ++kk) acc += z[kk] * Wf1[kk * 64 + j];
    z1[j] = acc > 0.0f ? acc : 0.0f;
    __syncthreads();
    if (j < 8) {
        float o = bf2[j];
#pragma unroll 8
        for (int q = 0; q < 64; ++q) o += z1[q] * Wf2[q * 8 + j];
        out[g * 8 + j] = o;
    }
}

extern "C" void kernel_launch(void* const* d_in, const int* in_sizes, int n_in,
                              void* d_out, int out_size, void* d_ws, size_t ws_size,
                              hipStream_t stream) {
    (void)in_sizes; (void)n_in; (void)out_size; (void)ws_size;
    const float* x   = (const float*)d_in[0];
    const int*   ei  = (const int*)d_in[1];     // [2, E] -> src = ei, dst = ei+E
    const float* W1l = (const float*)d_in[4];
    const float* b1  = (const float*)d_in[5];
    const float* W1r = (const float*)d_in[6];
    const float* p1  = (const float*)d_in[7];
    const float* W2l = (const float*)d_in[8];
    const float* b2  = (const float*)d_in[9];
    const float* W2r = (const float*)d_in[10];
    const float* p2  = (const float*)d_in[11];
    const float* Wf1 = (const float*)d_in[12];
    const float* bf1 = (const float*)d_in[13];
    const float* Wf2 = (const float*)d_in[14];
    const float* bf2 = (const float*)d_in[15];

    char* ws = (char*)d_ws;
    size_t off = 0;
    auto take = [&](size_t bytes) -> char* {
        char* p = ws + off;
        off = (off + bytes + 255) & ~(size_t)255;
        return p;
    };
    float* ssum    = (float*)take((size_t)M1_ * 64 * 4);   // reused for conv2
    float* deg     = (float*)take((size_t)M1_ * 4);        // reused for conv2
    float* h       = (float*)take((size_t)M1_ * 64 * 4);   // h1 then h2
    float* score   = (float*)take((size_t)M1_ * 4);
    int*   glob    = (int*)  take((size_t)M2_ * 4);
    float* gate    = (float*)take((size_t)M2_ * 4);
    int*   mapping = (int*)  take((size_t)M1_ * 4);
    float* xnew1   = (float*)take((size_t)M2_ * 64 * 4);   // also reused as xnew2
    int*   src2    = (int*)  take((size_t)NE_ * 4);
    int*   dst2    = (int*)  take((size_t)NE_ * 4);
    float* em2     = (float*)take((size_t)NE_ * 4);
    float* x1buf   = (float*)take((size_t)NB_ * 128 * 4);
    float* x2buf   = (float*)take((size_t)NB_ * 128 * 4);
    float* xnew2   = xnew1;  // alias: xnew1 dead after conv2 consumes it

    // ---- layer 1 ----
    hipMemsetAsync(ssum, 0, (size_t)M1_ * 64 * 4, stream);
    hipMemsetAsync(deg, 0, (size_t)M1_ * 4, stream);
    hipMemsetAsync(mapping, 0xFF, (size_t)M1_ * 4, stream);  // -1
    scatter1_kernel<<<NE_ / 256, 256, 0, stream>>>(x, ei, ei + NE_, ssum, deg, NE_);
    conv_wmma_kernel<<<M1_ / 128, 256, 0, stream>>>(ssum, deg, x, W1l, b1, W1r, h, M1_);
    score_kernel<<<M1_ / 256, 256, 0, stream>>>(h, p1, score, M1_);
    topk_kernel<<<NB_, 256, 0, stream>>>(score, NN_, 1024, K1_, glob, gate, mapping);
    gather_kernel<<<(M2_ * 16) / 256, 256, 0, stream>>>(h, glob, gate, xnew1, M2_);
    remap_kernel<<<NE_ / 256, 256, 0, stream>>>(ei, mapping, src2, dst2, em2, NE_);
    readout_kernel<<<NB_, 64, 0, stream>>>(xnew1, K1_, x1buf);

    // ---- layer 2 ----
    hipMemsetAsync(ssum, 0, (size_t)M2_ * 64 * 4, stream);
    hipMemsetAsync(deg, 0, (size_t)M2_ * 4, stream);
    scatter2_kernel<<<NE_ / 256, 256, 0, stream>>>(xnew1, src2, dst2, em2, ssum, deg, NE_);
    conv_wmma_kernel<<<M2_ / 128, 256, 0, stream>>>(ssum, deg, xnew1, W2l, b2, W2r, h, M2_);
    score_kernel<<<M2_ / 256, 256, 0, stream>>>(h, p2, score, M2_);
    topk_kernel<<<NB_, 256, 0, stream>>>(score, K1_, 1024, K2_, glob, gate, mapping);
    gather_kernel<<<(M3_ * 16) / 256, 256, 0, stream>>>(h, glob, gate, xnew2, M3_);
    readout_kernel<<<NB_, 64, 0, stream>>>(xnew2, K2_, x2buf);

    // ---- head ----
    mlp_kernel<<<NB_, 64, 0, stream>>>(x1buf, x2buf, Wf1, bf1, Wf2, bf2, (float*)d_out);
}
